// IAM_47794396069954
// MI455X (gfx1250) — matrix operand order repaired and checked by reading, not compiled, not run on hardware
//
#include <hip/hip_runtime.h>
#include <stdint.h>

#define BDIM 2
#define CCH  64
#define HDIM 320
#define WDIM 320

typedef __bf16 bf16_t;
typedef __attribute__((ext_vector_type(16))) __bf16 v16bf;
typedef __attribute__((ext_vector_type(8)))  float  v8f;

// ---------------------------------------------------------------------------
// WMMA fragment helpers (CDNA5 16x16x32 bf16, wave32 layouts per ISA 7.12.2)
// A (16x32, MxK): lane = hf*16+m holds row m; regs 0-3 -> K = hf*8+0..7,
//                 regs 4-7 -> K = 16+hf*8+0..7  (two contiguous 8-elem runs)
// B (32x16, KxN): lane = hf*16+n holds col n; same K order as A (two 8-elem runs)
// C/D (16x16 f32): elem r <-> M = hf*8+r, N = lane&15
// ---------------------------------------------------------------------------
__device__ __forceinline__ v16bf ld_a(const bf16_t* base, int stride, int lane, int koff) {
  int m = lane & 15, hf = (lane >> 4) & 1;
  const bf16_t* p0 = base + (long)m * stride + koff + hf * 8;
  v16bf r;
#pragma unroll
  for (int i = 0; i < 8; ++i) { r[i] = p0[i]; r[8 + i] = p0[16 + i]; }
  return r;
}

__device__ __forceinline__ v16bf ld_b(const bf16_t* base, int stride, int lane, int koff) {
  int n = lane & 15, hf = (lane >> 4) & 1;
  const bf16_t* p = base + (long)n * stride + koff + hf * 8;
  v16bf r;
#pragma unroll
  for (int i = 0; i < 8; ++i) { r[i] = p[i]; r[8 + i] = p[16 + i]; }
  return r;
}

__device__ __forceinline__ v8f wmma_bf16(v16bf a, v16bf b, v8f c) {
  return __builtin_amdgcn_wmma_f32_16x16x32_bf16(false, a, false, b, (short)0, c,
                                                 false, false);
}

// ---------------------------------------------------------------------------
// P0: pass-through: u,d (fp32 NCHW) -> low 64 channels of channels-last bf16
//     conv inputs xu=[b][h][w][128], xd=[b][h][w][128]
// ---------------------------------------------------------------------------
__global__ void __launch_bounds__(256) k_pass(const float* __restrict__ u,
                                              const float* __restrict__ d,
                                              bf16_t* __restrict__ xu,
                                              bf16_t* __restrict__ xd) {
  long i = (long)blockIdx.x * blockDim.x + threadIdx.x;
  const long N = (long)BDIM * CCH * HDIM * WDIM;
  if (i >= N) return;
  int w = (int)(i % WDIM); long t = i / WDIM;
  int h = (int)(t % HDIM); t /= HDIM;
  int c = (int)(t % CCH);  int b = (int)(t / CCH);
  long o = (((long)(b * HDIM + h) * WDIM) + w) * 128 + c;
  xu[o] = (bf16_t)u[i];
  xd[o] = (bf16_t)d[i];
}

// ---------------------------------------------------------------------------
// P1: 1x1 projections via WMMA.  One workgroup per (stream,b,h) row.
//     Input row staged in LDS as bf16 [w][c] (stride 66 to dodge bank
//     conflicts); both 64x64 weight matrices live in REGISTERS (16 fragments
//     = 128 VGPRs) for the whole kernel -- no spills, no per-tile B loads.
// ---------------------------------------------------------------------------
#define ROWP 66   // padded [w][c] row stride

__global__ void __launch_bounds__(256, 1)
k_proj(const float* __restrict__ u, const float* __restrict__ d,
       const float* __restrict__ Wu1, const float* __restrict__ Wu2,
       const float* __restrict__ Wd1, const float* __restrict__ Wd2,
       bf16_t* __restrict__ u1c, bf16_t* __restrict__ u2c,
       bf16_t* __restrict__ d1c, bf16_t* __restrict__ d2c) {
  extern __shared__ char smem[];
  bf16_t* wq   = (bf16_t*)smem;       // [64][64] row-major [cout][cin]
  bf16_t* wk   = wq + 64 * 64;
  bf16_t* rowl = wk + 64 * 64;        // [320][ROWP] input row, [w][c]

  int bx = blockIdx.x;
  int s = bx / (BDIM * HDIM);
  int rem = bx % (BDIM * HDIM);
  int b = rem / HDIM, h = rem % HDIM;

  const float* x  = s ? d   : u;
  const float* Wq = s ? Wd1 : Wu1;
  const float* Wk = s ? Wd2 : Wu2;
  bf16_t* qo = s ? d1c : u1c;
  bf16_t* ko = s ? d2c : u2c;

  int tid = threadIdx.x;
  const long cstride = (long)HDIM * WDIM;
  const long xbase   = (long)b * CCH * cstride + (long)h * WDIM;

  for (int j = tid; j < 64 * 64; j += blockDim.x) {
    wq[j] = (bf16_t)Wq[j];
    wk[j] = (bf16_t)Wk[j];
  }
  // stage the input row, transposed to [w][c] bf16 (reads coalesced along w)
  for (int j = tid; j < CCH * WDIM; j += blockDim.x) {
    int c = j / WDIM, w = j % WDIM;
    rowl[w * ROWP + c] = (bf16_t)x[xbase + (long)c * cstride + w];
  }
  __syncthreads();

  int wave = tid >> 5, lane = tid & 31, nw = blockDim.x >> 5;
  int hf = (lane >> 4) & 1, n = lane & 15;

  // register-resident weight fragments: 4 cout tiles x 2 K-chunks x 2 matrices
  v16bf Bq[4][2], Bk[4][2];
#pragma unroll
  for (int nt = 0; nt < 4; ++nt)
#pragma unroll
    for (int kk = 0; kk < 2; ++kk) {
      Bq[nt][kk] = ld_b(wq + nt * 16 * 64, 64, lane, kk * 32);
      Bk[nt][kk] = ld_b(wk + nt * 16 * 64, 64, lane, kk * 32);
    }

  const long rowout = (long)(b * HDIM + h) * WDIM * 64;
  for (int wt = wave; wt < WDIM / 16; wt += nw) {
    int w0 = wt * 16;
    v16bf a0 = ld_a(rowl + (long)w0 * ROWP, ROWP, lane, 0);
    v16bf a1 = ld_a(rowl + (long)w0 * ROWP, ROWP, lane, 32);
#pragma unroll
    for (int nt = 0; nt < 4; ++nt) {
      v8f cq = {}, ck = {};
      cq = wmma_bf16(a0, Bq[nt][0], cq);
      cq = wmma_bf16(a1, Bq[nt][1], cq);
      ck = wmma_bf16(a0, Bk[nt][0], ck);
      ck = wmma_bf16(a1, Bk[nt][1], ck);
#pragma unroll
      for (int r = 0; r < 8; ++r) {
        long pix = rowout + (long)(w0 + hf * 8 + r) * 64 + nt * 16 + n;
        qo[pix] = (bf16_t)cq[r];
        ko[pix] = (bf16_t)ck[r];
      }
    }
  }
}

// ---------------------------------------------------------------------------
// P2: weight repack  [O][I][3][3] fp32 -> [tap][O][I] bf16
// ---------------------------------------------------------------------------
__global__ void __launch_bounds__(256) k_wpack(const float* __restrict__ src,
                                               bf16_t* __restrict__ dst, int O, int I) {
  int n = blockIdx.x * blockDim.x + threadIdx.x;
  int tot = O * I * 9;
  if (n >= tot) return;
  int tap = n % 9;
  int i   = (n / 9) % I;
  int o   = n / (9 * I);
  dst[((long)tap * O + o) * I + i] = (bf16_t)src[n];
}

// ---------------------------------------------------------------------------
// ATT: axial attention per (stream,b,h) row.  512 threads (16 waves).
//   LDS: S[320][320] bf16 (200KB) + Q[320][64] (40KB) + K/V region (40KB).
//   GEMM1 (scores, K=64) -> softmax -> GEMM2 (apply, K=320), writes the
//   buffer into the high 64 channels of the channels-last conv input.
// ---------------------------------------------------------------------------
__global__ void __launch_bounds__(512, 1)
k_attn(const bf16_t* __restrict__ u1c, const bf16_t* __restrict__ u2c,
       const bf16_t* __restrict__ d1c, const bf16_t* __restrict__ d2c,
       const float* __restrict__ u, const float* __restrict__ d,
       bf16_t* __restrict__ xu, bf16_t* __restrict__ xd) {
  extern __shared__ char smem[];
  bf16_t* S  = (bf16_t*)smem;          // [320][320] scores (bf16)
  bf16_t* QL = S + WDIM * WDIM;        // [320][64]  queries, [w][c]
  bf16_t* KV = QL + WDIM * 64;         // phase1: keys [w][c]; phase2: V [c][w]

  int bx = blockIdx.x;
  int s = bx / (BDIM * HDIM);
  int rem = bx % (BDIM * HDIM);
  int b = rem / HDIM, h = rem % HDIM;

  const bf16_t* qsrc = s ? d1c : u1c;
  const bf16_t* ksrc = s ? u2c : d2c;
  const float*  vsrc = s ? u : d;
  bf16_t* xout = s ? xd : xu;

  int tid = threadIdx.x;
  const long rowq = (long)(b * HDIM + h) * WDIM * 64;
  const uint32_t* qs32 = (const uint32_t*)(qsrc + rowq);
  const uint32_t* ks32 = (const uint32_t*)(ksrc + rowq);
  uint32_t* QL32 = (uint32_t*)QL;
  uint32_t* KV32 = (uint32_t*)KV;
  for (int j = tid; j < WDIM * 64 / 2; j += blockDim.x) {
    QL32[j] = qs32[j];
    KV32[j] = ks32[j];
  }
  __syncthreads();

  int wave = tid >> 5, lane = tid & 31, nw = blockDim.x >> 5;
  int hf = (lane >> 4) & 1, n = lane & 15;

  // GEMM1: S[wq][wv] = sum_c Q[wq][c] * K[wv][c]     (400 tiles of 16x16)
  for (int t = wave; t < (WDIM / 16) * (WDIM / 16); t += nw) {
    int ti = t / (WDIM / 16), tj = t % (WDIM / 16);
    v8f acc = {};
#pragma unroll
    for (int kk = 0; kk < 2; ++kk) {
      v16bf a  = ld_a(QL + ti * 16 * 64, 64, lane, kk * 32);
      v16bf bb = ld_b(KV + tj * 16 * 64, 64, lane, kk * 32);
      acc = wmma_bf16(a, bb, acc);
    }
#pragma unroll
    for (int r = 0; r < 8; ++r)
      S[(ti * 16 + hf * 8 + r) * WDIM + tj * 16 + n] = (bf16_t)acc[r];
  }
  __syncthreads();

  // Load V[c][w] into KV region (keys no longer needed), softmax S rows.
  const long vbase = (long)b * CCH * HDIM * WDIM + (long)h * WDIM;
  for (int j = tid; j < CCH * WDIM; j += blockDim.x) {
    int c = j / WDIM, w = j % WDIM;
    KV[c * WDIM + w] = (bf16_t)vsrc[vbase + (long)c * HDIM * WDIM + w];
  }
  for (int row = tid; row < WDIM; row += blockDim.x) {
    bf16_t* Sr = S + row * WDIM;
    float mx = -3.4e38f;
    for (int j = 0; j < WDIM; ++j) { float v = (float)Sr[j]; mx = v > mx ? v : mx; }
    float sum = 0.f;
    for (int j = 0; j < WDIM; ++j) {
      float e = __expf((float)Sr[j] - mx);
      sum += e;
      Sr[j] = (bf16_t)e;
    }
    float inv = 1.f / sum;
    for (int j = 0; j < WDIM; ++j) Sr[j] = (bf16_t)((float)Sr[j] * inv);
  }
  __syncthreads();

  // GEMM2: buf[wq][c] = sum_v S[wq][v] * V[c][v]   (20 x 4 tiles, K=320)
  const long rowx = (long)(b * HDIM + h) * WDIM * 128;
  for (int t = wave; t < (WDIM / 16) * 4; t += nw) {
    int ti = t / 4, tn = t % 4;
    v8f acc = {};
#pragma unroll
    for (int kk = 0; kk < WDIM / 32; ++kk) {
      v16bf a  = ld_a(S + ti * 16 * WDIM, WDIM, lane, kk * 32);
      v16bf bb = ld_b(KV + tn * 16 * WDIM, WDIM, lane, kk * 32);
      acc = wmma_bf16(a, bb, acc);
    }
#pragma unroll
    for (int r = 0; r < 8; ++r)
      xout[rowx + (long)(ti * 16 + hf * 8 + r) * 128 + 64 + tn * 16 + n] = (bf16_t)acc[r];
  }
}

// ---------------------------------------------------------------------------
// 3x3 conv as implicit GEMM.  One workgroup per (b,h) output row, 8 waves.
// Weights staged in LDS as [tap][cout][cin] bf16.  Each wave owns one cout
// tile (nt = wave&3); for CIN==64 its 18 weight fragments (144 VGPRs) are
// preloaded into registers for the whole row.  Per 16-pixel x 16-cout tile:
// 9 taps x CIN/32 bf16 WMMAs with zero-padded boundary fragments.
// ---------------------------------------------------------------------------
template <int CIN, bool RELU, bool F32OUT>
__global__ void __launch_bounds__(256, 1)
k_conv3x3(const bf16_t* __restrict__ x, const bf16_t* __restrict__ wp,
          bf16_t* __restrict__ ob, float* __restrict__ of) {
  extern __shared__ char smem[];
  bf16_t* wl = (bf16_t*)smem;   // [9][64][CIN]

  int bx = blockIdx.x;
  int b = bx / HDIM, h = bx % HDIM;
  int tid = threadIdx.x;

  const int WCNT = 9 * 64 * CIN / 2;
  const uint32_t* ws32 = (const uint32_t*)wp;
  uint32_t* wl32 = (uint32_t*)wl;
  for (int j = tid; j < WCNT; j += blockDim.x) wl32[j] = ws32[j];
  __syncthreads();

  int wave = tid >> 5, lane = tid & 31, nw = blockDim.x >> 5;
  int m = lane & 15, hf = (lane >> 4) & 1, n = m;

  int nt = wave & 3;                 // fixed cout tile per wave
  int wstart = wave >> 2, wstep = nw >> 2;
  const long rowbase = (long)(b * HDIM + h) * WDIM;

  if (CIN == 64) {
    // register-resident weight fragments: 9 taps x 2 K-chunks
    v16bf B[9][2];
#pragma unroll
    for (int tap = 0; tap < 9; ++tap)
#pragma unroll
      for (int kk = 0; kk < 2; ++kk)
        B[tap][kk] = ld_b(wl + (tap * 64 + nt * 16) * CIN, CIN, lane, kk * 32);

    for (int wt = wstart; wt < WDIM / 16; wt += wstep) {
      int w0 = wt * 16;
      v8f acc = {};
#pragma unroll
      for (int dy = 0; dy < 3; ++dy) {
        int hh = h + dy - 1;
        if (hh < 0 || hh >= HDIM) continue;
#pragma unroll
        for (int dx = 0; dx < 3; ++dx) {
          int ww = w0 + m + dx - 1;
          bool pv = (ww >= 0) && (ww < WDIM);
#pragma unroll
          for (int kk = 0; kk < 2; ++kk) {
            v16bf a = {};
            if (pv) {
              const bf16_t* p0 =
                  x + ((long)(b * HDIM + hh) * WDIM + ww) * CIN + kk * 32 + hf * 8;
#pragma unroll
              for (int i = 0; i < 8; ++i) { a[i] = p0[i]; a[8 + i] = p0[16 + i]; }
            }
            acc = wmma_bf16(a, B[dy * 3 + dx][kk], acc);
          }
        }
      }
      if (F32OUT) {
#pragma unroll
        for (int r = 0; r < 8; ++r)
          of[(((long)(b * 64 + nt * 16 + n)) * HDIM + h) * WDIM + w0 + hf * 8 + r] = acc[r];
      } else {
#pragma unroll
        for (int r = 0; r < 8; ++r) {
          float v = acc[r];
          if (RELU) v = v > 0.f ? v : 0.f;
          ob[(rowbase + w0 + hf * 8 + r) * 64 + nt * 16 + n] = (bf16_t)v;
        }
      }
    }
  } else {
    for (int wt = wstart; wt < WDIM / 16; wt += wstep) {
      int w0 = wt * 16;
      {
        int wn = w0 + 16 < WDIM ? w0 + 16 : w0;
        __builtin_prefetch(x + (rowbase + wn + m) * CIN, 0, 1);
      }
      v8f acc = {};
      for (int dy = 0; dy < 3; ++dy) {
        int hh = h + dy - 1;
        if (hh < 0 || hh >= HDIM) continue;
        for (int dx = 0; dx < 3; ++dx) {
          int ww = w0 + m + dx - 1;
          bool pv = (ww >= 0) && (ww < WDIM);
          const bf16_t* bp = wl + ((dy * 3 + dx) * 64 + nt * 16) * CIN;
#pragma unroll
          for (int kk = 0; kk < CIN / 32; ++kk) {
            v16bf a = {};
            if (pv) {
              const bf16_t* p0 =
                  x + ((long)(b * HDIM + hh) * WDIM + ww) * CIN + kk * 32 + hf * 8;
#pragma unroll
              for (int i = 0; i < 8; ++i) { a[i] = p0[i]; a[8 + i] = p0[16 + i]; }
            }
            v16bf bb = ld_b(bp, CIN, lane, kk * 32);
            acc = wmma_bf16(a, bb, acc);
          }
        }
      }
      if (F32OUT) {
#pragma unroll
        for (int r = 0; r < 8; ++r)
          of[(((long)(b * 64 + nt * 16 + n)) * HDIM + h) * WDIM + w0 + hf * 8 + r] = acc[r];
      } else {
#pragma unroll
        for (int r = 0; r < 8; ++r) {
          float v = acc[r];
          if (RELU) v = v > 0.f ? v : 0.f;
          ob[(rowbase + w0 + hf * 8 + r) * 64 + nt * 16 + n] = (bf16_t)v;
        }
      }
    }
  }
}

// ---------------------------------------------------------------------------
extern "C" void kernel_launch(void* const* d_in, const int* in_sizes, int n_in,
                              void* d_out, int out_size, void* d_ws, size_t ws_size,
                              hipStream_t stream) {
  const float* u   = (const float*)d_in[0];
  const float* d   = (const float*)d_in[1];
  const float* Wu1 = (const float*)d_in[2];
  const float* Wu2 = (const float*)d_in[3];
  const float* Wd1 = (const float*)d_in[4];
  const float* Wd2 = (const float*)d_in[5];
  const float* W1a = (const float*)d_in[6];
  const float* W1b = (const float*)d_in[7];
  const float* W2a = (const float*)d_in[8];
  const float* W2b = (const float*)d_in[9];
  float* out = (float*)d_out;

  const long NC = (long)BDIM * CCH * HDIM * WDIM;  // 13,107,200 elems / stream

  char* ws = (char*)d_ws;
  bf16_t* u1c  = (bf16_t*)ws; ws += NC * 2;
  bf16_t* u2c  = (bf16_t*)ws; ws += NC * 2;
  bf16_t* d1c  = (bf16_t*)ws; ws += NC * 2;
  bf16_t* d2c  = (bf16_t*)ws; ws += NC * 2;
  bf16_t* xu   = (bf16_t*)ws; ws += NC * 4;   // [b][h][w][128]
  bf16_t* xd   = (bf16_t*)ws; ws += NC * 4;
  bf16_t* mu   = (bf16_t*)ws; ws += NC * 2;   // [b][h][w][64]
  bf16_t* md   = (bf16_t*)ws; ws += NC * 2;
  bf16_t* wp1a = (bf16_t*)ws; ws += 9 * 64 * 128 * 2;
  bf16_t* wp1b = (bf16_t*)ws; ws += 9 * 64 * 64 * 2;
  bf16_t* wp2a = (bf16_t*)ws; ws += 9 * 64 * 128 * 2;
  bf16_t* wp2b = (bf16_t*)ws; ws += 9 * 64 * 64 * 2;

  // 1. pass-through channels into channels-last conv inputs
  k_pass<<<(int)((NC + 255) / 256), 256, 0, stream>>>(u, d, xu, xd);

  // 2. 1x1 projections (WMMA, register-resident weights)
  size_t pjlds = (size_t)(2 * 64 * 64 + WDIM * ROWP) * sizeof(bf16_t);
  k_proj<<<2 * BDIM * HDIM, 256, pjlds, stream>>>(u, d, Wu1, Wu2, Wd1, Wd2,
                                                  u1c, u2c, d1c, d2c);

  // 3. weight repack
  k_wpack<<<(64 * 128 * 9 + 255) / 256, 256, 0, stream>>>(W1a, wp1a, 64, 128);
  k_wpack<<<(64 * 64 * 9 + 255) / 256, 256, 0, stream>>>(W1b, wp1b, 64, 64);
  k_wpack<<<(64 * 128 * 9 + 255) / 256, 256, 0, stream>>>(W2a, wp2a, 64, 128);
  k_wpack<<<(64 * 64 * 9 + 255) / 256, 256, 0, stream>>>(W2b, wp2b, 64, 64);

  // 4. axial attention (both streams), 280KB dynamic LDS per WGP
  size_t attlds = (size_t)(WDIM * WDIM + 2 * WDIM * 64) * sizeof(bf16_t);
  k_attn<<<2 * BDIM * HDIM, 512, attlds, stream>>>(u1c, u2c, d1c, d2c, u, d, xu, xd);

  // 5. conv3x3(128->64) + relu
  size_t c1lds = 9 * 64 * 128 * sizeof(bf16_t);
  k_conv3x3<128, true, false><<<BDIM * HDIM, 256, c1lds, stream>>>(xu, wp1a, mu, nullptr);
  k_conv3x3<128, true, false><<<BDIM * HDIM, 256, c1lds, stream>>>(xd, wp2a, md, nullptr);

  // 6. conv3x3(64->64) -> fp32 NCHW outputs
  size_t c2lds = 9 * 64 * 64 * sizeof(bf16_t);
  k_conv3x3<64, false, true><<<BDIM * HDIM, 256, c2lds, stream>>>(mu, wp1b, nullptr, out);
  k_conv3x3<64, false, true><<<BDIM * HDIM, 256, c2lds, stream>>>(md, wp2b, nullptr, out + NC);
}
